// PQRQuantizer_29970281791599
// MI455X (gfx1250) — matrix operand order repaired and checked by reading, not compiled, not run on hardware
//
#include <hip/hip_runtime.h>
#include <hip/hip_bf16.h>

// ---------------------------------------------------------------------------
// VQ codebook search on gfx1250:
//   indices[n] = argmin_k ( c2[k] - 2 * <inputs[n], codebook[k]> )
//   quantized  = inputs   (straight-through: q + (x - q) == x)
// Compute-bound (AI ~1900 flop/byte) -> bf16 WMMA with f32 accumulation.
// v3: A panel in registers + B fragments software-pipelined one K-chunk
//     ahead in a register double buffer, so WMMAs don't stall on dscnt 0.
// ---------------------------------------------------------------------------

typedef __attribute__((ext_vector_type(16))) __bf16 v16bf;
typedef __attribute__((ext_vector_type(8)))  float  v8f;

#define N_TOK        32768
#define DIM          512
#define KCODES       8192
#define ROWS_PER_WG  128
#define TILE_CODES   64
#define KCHUNKS      (DIM / 32)                  // 16
#define LDS_STRIDE   520                         // bf16 elems per row (512 + 8 pad)
#define LDS_ROW_B    (LDS_STRIDE * 2)            // 1040 bytes
#define A_BYTES      (ROWS_PER_WG * LDS_ROW_B)   // 133120
#define CB_TILE_B    (TILE_CODES * LDS_ROW_B)    // 66560
#define LDS_TOTAL    (A_BYTES + 2 * CB_TILE_B)   // 266240 <= 320KB/WGP

union Frag {
    uint4 q[2];
    v16bf v;
};

static __device__ inline unsigned f2bf1(float f) {
    // round-to-nearest-even f32 -> bf16 bit pattern
    unsigned u = __float_as_uint(f);
    return (u + 0x7FFFu + ((u >> 16) & 1u)) >> 16;
}

// ---------------------------------------------------------------------------
// Kernel 1: codebook f32 -> bf16 (workspace) + per-code squared norm c2[k]
// ---------------------------------------------------------------------------
__global__ __launch_bounds__(256)
void vq_prep_codebook(const float* __restrict__ cb,
                      unsigned* __restrict__ cb_bf_u32,   // packed 2x bf16
                      float* __restrict__ c2) {
    __shared__ float partial[8];
    const unsigned row = blockIdx.x;       // one code row per block
    const unsigned tid = threadIdx.x;      // 256 threads, 2 elems each
    const float2 v = ((const float2*)(cb + (size_t)row * DIM))[tid];
    float s = v.x * v.x + v.y * v.y;
    cb_bf_u32[row * (DIM / 2) + tid] = f2bf1(v.x) | (f2bf1(v.y) << 16);
#pragma unroll
    for (int m = 16; m >= 1; m >>= 1) s += __shfl_xor(s, m, 32);
    if ((tid & 31u) == 0u) partial[tid >> 5] = s;
    __syncthreads();
    if (tid == 0) {
        float t = 0.f;
#pragma unroll
        for (int w = 0; w < 8; ++w) t += partial[w];
        c2[row] = t;
    }
}

// ---------------------------------------------------------------------------
// Kernel 2: fused distance + argmin.
// Grid: 256 WGs x 256 thr (8 waves). WG owns 128 rows; wave w owns rows
// [16w, 16w+16) and keeps its whole 16x512 bf16 A panel in registers.
// Codebook streamed in 64-code tiles, double-buffered via async global->LDS
// DMA (ASYNCcnt). Per tile per wave: 16 K-chunks x 4 code-subtiles =
// 64 x v_wmma_f32_16x16x32_bf16 into 4 independent accumulators, with the
// B fragments register-double-buffered one K-chunk ahead.
// ---------------------------------------------------------------------------
__global__ __launch_bounds__(256, 1)
void vq_argmin_kernel(const float* __restrict__ inputs,
                      const unsigned short* __restrict__ cb_bf,
                      const float* __restrict__ c2,
                      float* __restrict__ idx_out) {
    extern __shared__ char smem[];
    char* smemA = smem;                       // 128 x 520 bf16
    const unsigned tid    = threadIdx.x;
    const unsigned lane   = tid & 31u;
    const unsigned wave   = tid >> 5;         // 0..7
    const unsigned lane15 = lane & 15u;
    const unsigned laneHi = lane >> 4;        // 0 or 1
    const unsigned row0   = blockIdx.x * ROWS_PER_WG;

    // ---- Stage A: convert this WG's 128 input rows f32 -> bf16 in LDS ----
    {
        const float4* src = (const float4*)(inputs + (size_t)row0 * DIM);
        for (unsigned i = tid; i < ROWS_PER_WG * (DIM / 4); i += 256u) {
            const unsigned r = i >> 7, c4 = i & 127u;   // DIM/4 == 128
            const float4 x = src[r * (DIM / 4) + c4];
            uint2 p;
            p.x = f2bf1(x.x) | (f2bf1(x.y) << 16);
            p.y = f2bf1(x.z) | (f2bf1(x.w) << 16);
            *(uint2*)(smemA + r * LDS_ROW_B + c4 * 8u) = p;
        }
    }

    const unsigned ldsBBase = (unsigned)(size_t)(smem + A_BYTES);

    // Async DMA of one 64-code bf16 tile into LDS buffer b (tracked by ASYNCcnt)
    auto prefetch = [&](unsigned t, unsigned b) {
        const unsigned bufBase = ldsBBase + b * CB_TILE_B;
#pragma unroll
        for (unsigned j = 0; j < 16; ++j) {           // 4096 x 16B chunks / 256 thr
            const unsigned c   = tid + 256u * j;
            const unsigned r   = c >> 6, c16 = c & 63u;
            const unsigned lds = bufBase + r * LDS_ROW_B + c16 * 16u;
            const unsigned off = (t * TILE_CODES + r) * (DIM * 2u) + c16 * 16u;
            asm volatile("global_load_async_to_lds_b128 %0, %1, %2 offset:0"
                         :: "v"(lds), "v"(off), "s"(cb_bf) : "memory");
        }
    };

    prefetch(0u, 0u);
    asm volatile("s_wait_asynccnt 0" ::: "memory");
    __syncthreads();                                   // A + first tile ready

    // ---- Load this wave's entire A panel into registers (16 frags) ----
    // ISA 16-bit A layout: lane<16 -> row m=lane, K {0..7} then {16..23};
    // lane>=16 -> K {8..15} then {24..31}. Two 16B LDS reads, 32B apart.
    const unsigned aRowOff = (wave * 16u + lane15) * LDS_ROW_B;
    Frag fa[KCHUNKS];
#pragma unroll
    for (unsigned kc = 0; kc < KCHUNKS; ++kc) {
        const char* ap = smemA + aRowOff + kc * 64u + laneHi * 16u;
        fa[kc].q[0] = *(const uint4*)(ap);
        fa[kc].q[1] = *(const uint4*)(ap + 32);
    }

    float    best[8];
    unsigned bidx[8];
#pragma unroll
    for (int r = 0; r < 8; ++r) { best[r] = __builtin_inff(); bidx[r] = 0u; }

    const unsigned numTiles = KCODES / TILE_CODES;     // 128
    for (unsigned t = 0; t < numTiles; ++t) {
        const unsigned buf = t & 1u;
        if (t + 1u < numTiles) prefetch(t + 1u, buf ^ 1u);

        const char* bBase = smem + A_BYTES + buf * CB_TILE_B;
        // B layout (32x16), matching A: lane<16 -> col n=lane, K {0..7} then {16..23};
        // lane>=16 -> K {8..15} then {24..31}. Two 16B reads, 32B apart.
        const char* bRow = bBase + lane15 * LDS_ROW_B + laneHi * 16u;

        v8f acc[4];
#pragma unroll
        for (int s = 0; s < 4; ++s) acc[s] = (v8f){};

        // Register double-buffer of B fragments, pipelined one K-chunk ahead.
        Frag fb[2][4];
#pragma unroll
        for (int s = 0; s < 4; ++s) {
            const char* bps = bRow + s * 16 * LDS_ROW_B;
            fb[0][s].q[0] = *(const uint4*)(bps);
            fb[0][s].q[1] = *(const uint4*)(bps + 32);
        }
#pragma unroll
        for (unsigned kc = 0; kc < KCHUNKS; ++kc) {    // 16 K-chunks of 32
            const unsigned cur = kc & 1u;
            const unsigned nxt = cur ^ 1u;
            if (kc + 1u < KCHUNKS) {                   // prefetch next K-chunk's B
                const char* bp = bRow + (kc + 1u) * 64u;
#pragma unroll
                for (int s = 0; s < 4; ++s) {
                    const char* bps = bp + s * 16 * LDS_ROW_B;
                    fb[nxt][s].q[0] = *(const uint4*)(bps);
                    fb[nxt][s].q[1] = *(const uint4*)(bps + 32);
                }
            }
#pragma unroll
            for (int s = 0; s < 4; ++s)
                acc[s] = __builtin_amdgcn_wmma_f32_16x16x32_bf16(
                    false, fa[kc].v, false, fb[cur][s].v, (short)0, acc[s],
                    false, false);
        }

        // Epilogue: d = c2[n] - 2*dot ; running per-row argmin.
        // C/D layout: lane<16 -> rows 0..7 in v[0..7]; lane>=16 -> rows 8..15.
#pragma unroll
        for (int s = 0; s < 4; ++s) {
            const unsigned n = t * TILE_CODES + s * 16u + lane15;
            const float c2n = c2[n];
#pragma unroll
            for (int r = 0; r < 8; ++r) {
                const float d = __builtin_fmaf(-2.0f, acc[s][r], c2n);
                if (d < best[r]) { best[r] = d; bidx[r] = n; }
            }
        }

        asm volatile("s_wait_asynccnt 0" ::: "memory");
        __syncthreads();                               // next buffer ready
    }

    // Cross-lane argmin over the 16 N-columns sharing the same row set.
#pragma unroll
    for (int r = 0; r < 8; ++r) {
        float    v = best[r];
        unsigned i = bidx[r];
#pragma unroll
        for (int m = 8; m >= 1; m >>= 1) {
            const float    ov = __shfl_xor(v, m, 32);
            const unsigned oi = __shfl_xor(i, m, 32);
            if (ov < v || (ov == v && oi < i)) { v = ov; i = oi; }
        }
        best[r] = v; bidx[r] = i;
    }
    if (lane15 == 0u) {                                // lane 0 and lane 16
        const unsigned rowBase = row0 + wave * 16u + laneHi * 8u;
#pragma unroll
        for (int r = 0; r < 8; ++r)
            idx_out[rowBase + r] = (float)bidx[r];
    }
}

// ---------------------------------------------------------------------------
// Kernel 3: quantized output == inputs (straight-through identity).
// ---------------------------------------------------------------------------
__global__ __launch_bounds__(256)
void vq_copy_inputs(const float4* __restrict__ in, float4* __restrict__ out,
                    unsigned n4) {
    unsigned i = blockIdx.x * blockDim.x + threadIdx.x;
    const unsigned stride = gridDim.x * blockDim.x;
    for (; i < n4; i += stride) out[i] = in[i];
}

// ---------------------------------------------------------------------------
extern "C" void kernel_launch(void* const* d_in, const int* in_sizes, int n_in,
                              void* d_out, int out_size, void* d_ws, size_t ws_size,
                              hipStream_t stream) {
    const float* inputs   = (const float*)d_in[0];
    const float* codebook = (const float*)d_in[1];
    float* out = (float*)d_out;

    // workspace: [0, 8MB) codebook bf16, then c2[K] f32
    unsigned short* cb_bf = (unsigned short*)d_ws;
    float* c2 = (float*)((char*)d_ws + (size_t)KCODES * DIM * 2);

    vq_prep_codebook<<<KCODES, 256, 0, stream>>>(codebook, (unsigned*)cb_bf, c2);

    vq_argmin_kernel<<<N_TOK / ROWS_PER_WG, 256, LDS_TOTAL, stream>>>(
        inputs, cb_bf, c2, out + (size_t)N_TOK * DIM);

    vq_copy_inputs<<<4096, 256, 0, stream>>>(
        (const float4*)inputs, (float4*)out, (unsigned)(N_TOK * DIM / 4));
}
